// SGNP_45028437131846
// MI455X (gfx1250) — matrix operand order repaired and checked by reading, not compiled, not run on hardware
//
#include <hip/hip_runtime.h>
#include <math.h>

// ---------------------------------------------------------------------------
// Problem constants (match reference)
// ---------------------------------------------------------------------------
constexpr int B_    = 16;
constexpr int N_C_  = 2048;
constexpr int N_T_  = 1024;
constexpr int NCTX  = B_ * N_C_;            // 32768 ctx nodes (node ids 0..NCTX-1)
constexpr int NTEST = B_ * N_T_;            // 16384 test nodes
constexpr int NTOT  = NCTX + NTEST;         // 49152 nodes
constexpr int K_    = 16;                   // neighbors
constexpr int H_    = 4;                    // heads

typedef __attribute__((ext_vector_type(16))) _Float16 v16h;
typedef __attribute__((ext_vector_type(8)))  _Float16 v8h;
typedef __attribute__((ext_vector_type(8)))  float    v8f;

// D = A(16x32 f16) * B(32x16 f16) + C(16x16 f32)
#define WMMA_F16(a, b, c) \
  __builtin_amdgcn_wmma_f32_16x16x32_f16(false, (a), false, (b), (short)0, (c), false, false)

// ---------------------------------------------------------------------------
// Fast transcendentals (gfx1250 has native v_tanh_f32; fall back to a
// branch-free exp2 formulation that saturates correctly via inf/0).
// ---------------------------------------------------------------------------
__device__ __forceinline__ float fast_tanh_f(float x) {
#if __has_builtin(__builtin_amdgcn_tanhf)
  return __builtin_amdgcn_tanhf(x);
#elif __has_builtin(__builtin_amdgcn_tanh_f32)
  return __builtin_amdgcn_tanh_f32(x);
#else
  // tanh(x) = 1 - 2/(exp(2x)+1); exp via single v_exp_f32 (2^y)
  const float e = __builtin_amdgcn_exp2f(x * 2.8853900817779268f); // 2*log2(e)
  return 1.0f - 2.0f / (e + 1.0f);
#endif
}

__device__ __forceinline__ float fast_exp_f(float x) {
  return __builtin_amdgcn_exp2f(x * 1.4426950408889634f);          // log2(e)
}

__device__ __forceinline__ float gelu_f(float x) {
  // jax.nn.gelu default (tanh approximation)
  const float kA = 0.7978845608028654f;
  return 0.5f * x * (1.0f + fast_tanh_f(kA * (x + 0.044715f * x * x * x)));
}

__device__ __forceinline__ float softplus_f(float x) {
  return fmaxf(x, 0.0f) + log1pf(fast_exp_f(-fabsf(x)));
}

// ---------------------------------------------------------------------------
// WMMA fragment loaders (ISA 7.12.2 16-bit layouts)
// ---------------------------------------------------------------------------
// A fragment, 16x32 f16, LDS row-major (ld halves per row).
// lane<16 holds M=lane, K {0..7,16..23}; lane>=16 K {8..15,24..31}.
__device__ __forceinline__ v16h load_A_f16(const _Float16* A, int ld) {
  const int l = threadIdx.x & 31;
  const _Float16* p = A + (l & 15) * ld + ((l >> 4) << 3);
  v8h lo = *(const v8h*)p;
  v8h hi = *(const v8h*)(p + 16);
  v16h a;
#pragma unroll
  for (int j = 0; j < 8; ++j) { a[j] = lo[j]; a[8 + j] = hi[j]; }
  return a;
}

// B fragment, 32x16 f16, LDS N-major (Bt[n*ld + k]).
// lane n = l&15; lanes 0-15 hold K {0..7,16..23}, lanes 16-31 hold K {8..15,24..31}.
__device__ __forceinline__ v16h load_B_f16(const _Float16* Bt, int ld) {
  const int l = threadIdx.x & 31;
  const _Float16* p = Bt + (l & 15) * ld + ((l >> 4) << 3);
  v8h lo = *(const v8h*)p;
  v8h hi = *(const v8h*)(p + 16);
  v16h b;
#pragma unroll
  for (int j = 0; j < 8; ++j) { b[j] = lo[j]; b[8 + j] = hi[j]; }
  return b;
}

// ---------------------------------------------------------------------------
// Weight-panel layout in workspace (halves), produced once by prep_weights:
//   W1t [256 x 32 ]  off 0        (K padded 8->32 with zeros)
//   W2t [128 x 256]  off 8192
//   W3t [ 64 x 128]  off 40960
//   Wgt [ 64 x 64 ]  off 49152
//   H1t [256 x 64 ]  off 53248
//   H2t [ 64 x 256]  off 69632    total 86016 halves
// ---------------------------------------------------------------------------
constexpr int WB_W1 = 0;
constexpr int WB_W2 = 8192;
constexpr int WB_W3 = 40960;
constexpr int WB_WG = 49152;
constexpr int WB_H1 = 53248;
constexpr int WB_H2 = 69632;
constexpr int WB_TOT = 86016;
constexpr int WENC_H = WB_H1;            // encoder panel: 53248 halves
constexpr int WHEAD_H = WB_TOT - WB_H1;  // head panel:    32768 halves

__global__ void __launch_bounds__(256) prep_weights(
    const float* __restrict__ W1, const float* __restrict__ W2,
    const float* __restrict__ W3, const float* __restrict__ Wg,
    const float* __restrict__ H1, const float* __restrict__ H2,
    _Float16* __restrict__ wbuf) {
  const int i = blockIdx.x * 256 + threadIdx.x;
  if (i >= WB_TOT) return;
  float v;
  if (i < WB_W2) {                    // W1t[n*32+k] = W1[k*256+n], k<8 else 0
    const int n = i >> 5, k = i & 31;
    v = (k < 8) ? W1[(size_t)k * 256 + n] : 0.0f;
  } else if (i < WB_W3) {             // W2t[n*256+k] = W2[k*128+n]
    const int j = i - WB_W2, n = j >> 8, k = j & 255;
    v = W2[(size_t)k * 128 + n];
  } else if (i < WB_WG) {             // W3t[n*128+k] = W3[k*64+n]
    const int j = i - WB_W3, n = j >> 7, k = j & 127;
    v = W3[(size_t)k * 64 + n];
  } else if (i < WB_H1) {             // Wgt[n*64+k] = Wg[k*64+n]
    const int j = i - WB_WG, n = j >> 6, k = j & 63;
    v = Wg[(size_t)k * 64 + n];
  } else if (i < WB_H2) {             // H1t[n*64+k] = H1[k*256+n]
    const int j = i - WB_H1, n = j >> 6, k = j & 63;
    v = H1[(size_t)k * 256 + n];
  } else {                            // H2t[n*256+k] = H2[k*64+n]
    const int j = i - WB_H2, n = j >> 8, k = j & 255;
    v = H2[(size_t)k * 64 + n];
  }
  wbuf[i] = (_Float16)v;
}

// ---------------------------------------------------------------------------
// Kernel 1: kNN (test vs ctx only; ctx-ctx edges are dead code for the output)
// ---------------------------------------------------------------------------
__global__ void __launch_bounds__(256) knn_kernel(
    const float* __restrict__ s_ctx, const float* __restrict__ s_test,
    const float* __restrict__ We, const float* __restrict__ be,
    int* __restrict__ nbr, float* __restrict__ ebias) {
  __shared__ float sc[N_C_ * 2];
  const int t = threadIdx.x;
  const int q0 = blockIdx.x * 256;          // 256 | N_T  -> block within one batch
  const int b  = q0 / N_T_;
  for (int i = t; i < N_C_ * 2; i += 256) sc[i] = s_ctx[(size_t)b * N_C_ * 2 + i];
  __syncthreads();

  const int q = q0 + t;
  const float qx = s_test[(size_t)q * 2 + 0];
  const float qy = s_test[(size_t)q * 2 + 1];

  float dist[K_]; int idr[K_];
#pragma unroll
  for (int j = 0; j < K_; ++j) { dist[j] = 3.4e38f; idr[j] = 0; }

  for (int c = 0; c < N_C_; ++c) {
    const float dx = qx - sc[2 * c], dy = qy - sc[2 * c + 1];
    float d = dx * dx + dy * dy;
    if (d < dist[K_ - 1]) {
      float dv = d; int iv = c;
#pragma unroll
      for (int j = 0; j < K_; ++j) {
        if (dv < dist[j]) {
          float td = dist[j]; int ti = idr[j];
          dist[j] = dv; idr[j] = iv; dv = td; iv = ti;
        }
      }
    }
  }

  float w[8], bb[4];
#pragma unroll
  for (int i = 0; i < 8; ++i) w[i] = We[i];    // We (2,4) row-major
#pragma unroll
  for (int h = 0; h < H_; ++h) bb[h] = be[h];

#pragma unroll
  for (int k = 0; k < K_; ++k) {
    const int ci = idr[k];
    nbr[(size_t)q * K_ + k] = b * N_C_ + ci;   // global ctx sender id
    const float dx = qx - sc[2 * ci], dy = qy - sc[2 * ci + 1];
#pragma unroll
    for (int h = 0; h < H_; ++h)
      ebias[(size_t)q * (K_ * H_) + k * H_ + h] = dx * w[h] + dy * w[4 + h] + bb[h];
  }
}

// ---------------------------------------------------------------------------
// Kernel 2: fused encoder: 3-layer MLP (WMMA) + LayerNorm + xp = ln @ Wg (WMMA)
// 128 threads = 4 waves, 64 rows/block. Pre-packed f16 weights -> LDS via b128.
// ---------------------------------------------------------------------------
constexpr size_t ENC_H =
    (size_t)WENC_H +     // W1t|W2t|W3t|Wgt contiguous panel
    (size_t)64 * 32 +    // A0 input tile
    (size_t)64 * 256 +   // act1
    (size_t)64 * 128 +   // act2
    (size_t)64 * 64;     // lnh (f16 LN output)
constexpr size_t ENC_SMEM = ENC_H * 2 + (size_t)(64 * 64 + 256 + 128 + 64 + 64 + 64) * 4;

__global__ void __launch_bounds__(128) encoder_kernel(
    const float* __restrict__ s_ctx, const float* __restrict__ f_ctx,
    const float* __restrict__ s_test, const float* __restrict__ emb,
    const _Float16* __restrict__ wbuf,
    const float* __restrict__ b1, const float* __restrict__ b2,
    const float* __restrict__ b3,
    const float* __restrict__ ln_g, const float* __restrict__ ln_b,
    float* __restrict__ nodes, float* __restrict__ xp) {
  extern __shared__ char smem[];
  _Float16* sW1  = (_Float16*)smem;            // 256 x 32  (N-major)
  _Float16* sW2  = sW1 + (WB_W2 - WB_W1);
  _Float16* sW3  = sW1 + (WB_W3 - WB_W1);
  _Float16* sWg  = sW1 + (WB_WG - WB_W1);
  _Float16* sA0  = sW1 + WENC_H;               // 64 x 32
  _Float16* sAc1 = sA0 + 64 * 32;              // 64 x 256
  _Float16* sAc2 = sAc1 + 64 * 256;            // 64 x 128
  _Float16* sLnh = sAc2 + 64 * 128;            // 64 x 64
  float* sLnIn = (float*)(sLnh + 64 * 64);     // 64 x 64 (pre-LN, f32)
  float* sB1 = sLnIn + 64 * 64;
  float* sB2 = sB1 + 256;
  float* sB3 = sB2 + 128;
  float* sG  = sB3 + 64;
  float* sBt = sG + 64;

  const int t = threadIdx.x;
  // --- stage pre-packed weights: contiguous 16B copies ---
  {
    const v8h* src = (const v8h*)wbuf;
    v8h* dst = (v8h*)sW1;
    for (int i = t; i < WENC_H / 8; i += 128) dst[i] = src[i];
  }
  for (int i = t; i < 256; i += 128) sB1[i] = b1[i];
  for (int i = t; i < 128; i += 128) sB2[i] = b2[i];
  if (t < 64) { sB3[t] = b3[t]; sG[t] = ln_g[t]; sBt[t] = ln_b[t]; }

  // --- stage 64 input rows: [emb_obs(type), s(2), f(2), pad->32] ---
  const int base = blockIdx.x * 64;
  for (int i = t; i < 64 * 32; i += 128) {
    const int r = i >> 5, k = i & 31;
    const int node = base + r;
    float v = 0.0f;
    if (k < 8) {
      if (node < NCTX) {
        const int bb = node / N_C_, ci = node % N_C_;
        if (k < 4)      v = emb[4 + k];                                   // emb_obs[1]
        else if (k < 6) v = s_ctx[((size_t)bb * N_C_ + ci) * 2 + (k - 4)];
        else            v = f_ctx[((size_t)bb * N_C_ + ci) * 2 + (k - 6)];
      } else {
        const int tn = node - NCTX, bb = tn / N_T_, ti = tn % N_T_;
        if (k < 4)      v = emb[k];                                       // emb_obs[0]
        else if (k < 6) v = s_test[((size_t)bb * N_T_ + ti) * 2 + (k - 4)];
        // f part is zeros for test
      }
    }
    sA0[i] = (_Float16)v;
  }
  __syncthreads();

  const int wave = t >> 5, l = t & 31;
  const int m0 = wave * 16, lane_n = l & 15, mbase = (l >> 4) << 3;

  // --- layer 1: 16x32 @ 32x256 -> gelu -> act1 ---
  {
    v16h a = load_A_f16(sA0 + m0 * 32, 32);
#pragma unroll
    for (int nt = 0; nt < 16; ++nt) {
      v8f c = {};
      c = WMMA_F16(a, load_B_f16(sW1 + (nt * 16) * 32, 32), c);
      const int n = nt * 16 + lane_n;
      const float bias = sB1[n];
#pragma unroll
      for (int v = 0; v < 8; ++v)
        sAc1[(m0 + mbase + v) * 256 + n] = (_Float16)gelu_f(c[v] + bias);
    }
  }
  __syncthreads();

  // --- layer 2: 16x256 @ 256x128 -> gelu -> act2 ---
  {
    v16h a1[8];
#pragma unroll
    for (int kc = 0; kc < 8; ++kc) a1[kc] = load_A_f16(sAc1 + m0 * 256 + kc * 32, 256);
#pragma unroll
    for (int nt = 0; nt < 8; ++nt) {
      v8f c = {};
#pragma unroll
      for (int kc = 0; kc < 8; ++kc)
        c = WMMA_F16(a1[kc], load_B_f16(sW2 + (nt * 16) * 256 + kc * 32, 256), c);
      const int n = nt * 16 + lane_n;
      const float bias = sB2[n];
#pragma unroll
      for (int v = 0; v < 8; ++v)
        sAc2[(m0 + mbase + v) * 128 + n] = (_Float16)gelu_f(c[v] + bias);
    }
  }
  __syncthreads();

  // --- layer 3: 16x128 @ 128x64 + b3 -> pre-LN f32 ---
  {
    v16h a2[4];
#pragma unroll
    for (int kc = 0; kc < 4; ++kc) a2[kc] = load_A_f16(sAc2 + m0 * 128 + kc * 32, 128);
#pragma unroll
    for (int nt = 0; nt < 4; ++nt) {
      v8f c = {};
#pragma unroll
      for (int kc = 0; kc < 4; ++kc)
        c = WMMA_F16(a2[kc], load_B_f16(sW3 + (nt * 16) * 128 + kc * 32, 128), c);
      const int n = nt * 16 + lane_n;
      const float bias = sB3[n];
#pragma unroll
      for (int v = 0; v < 8; ++v)
        sLnIn[(m0 + mbase + v) * 64 + n] = c[v] + bias;
    }
  }
  __syncthreads();

  // --- LayerNorm over F=64, one thread per row ---
  if (t < 64) {
    float mu = 0.0f;
#pragma unroll
    for (int c = 0; c < 64; ++c) mu += sLnIn[t * 64 + c];
    mu *= (1.0f / 64.0f);
    float s2 = 0.0f;
#pragma unroll
    for (int c = 0; c < 64; ++c) { float d = sLnIn[t * 64 + c] - mu; s2 += d * d; }
    const float inv = rsqrtf(s2 * (1.0f / 64.0f) + 1e-6f);
    const size_t node = (size_t)(base + t);
#pragma unroll
    for (int c = 0; c < 64; ++c) {
      const float y = (sLnIn[t * 64 + c] - mu) * inv * sG[c] + sBt[c];
      nodes[node * 64 + c] = y;
      sLnh[t * 64 + c] = (_Float16)y;
    }
  }
  __syncthreads();

  // --- xp = ln @ Wg : 16x64 @ 64x64 ---
  {
    v16h ax0 = load_A_f16(sLnh + m0 * 64, 64);
    v16h ax1 = load_A_f16(sLnh + m0 * 64 + 32, 64);
#pragma unroll
    for (int nt = 0; nt < 4; ++nt) {
      v8f c = {};
      c = WMMA_F16(ax0, load_B_f16(sWg + (nt * 16) * 64, 64), c);
      c = WMMA_F16(ax1, load_B_f16(sWg + (nt * 16) * 64 + 32, 64), c);
      const int n = nt * 16 + lane_n;
#pragma unroll
      for (int v = 0; v < 8; ++v)
        xp[(size_t)(base + m0 + mbase + v) * 64 + n] = c[v];
    }
  }
}

// ---------------------------------------------------------------------------
// Kernel 3: GAT attention + residual, test receivers only (one thread/node)
// ---------------------------------------------------------------------------
__global__ void __launch_bounds__(128) gat_kernel(
    const float* __restrict__ xp, const int* __restrict__ nbr,
    const float* __restrict__ ebias, const float* __restrict__ a_src,
    const float* __restrict__ a_dst, float* __restrict__ nodes) {
  __shared__ float s_src[64], s_dst[64];
  const int t = threadIdx.x;
  if (t < 64) { s_src[t] = a_src[t]; s_dst[t] = a_dst[t]; }
  __syncthreads();

  const int q = blockIdx.x * 128 + t;          // test node index
  const size_t n = (size_t)NCTX + q;
  const float* xs = xp + n * 64;

  float sd[H_];
#pragma unroll
  for (int h = 0; h < H_; ++h) {
    float acc = 0.0f;
#pragma unroll
    for (int d = 0; d < 16; ++d) acc += xs[h * 16 + d] * s_dst[h * 16 + d];
    sd[h] = acc;
  }

  float sc[H_][K_];
#pragma unroll
  for (int k = 0; k < K_; ++k) {
    const float* ms = xp + (size_t)nbr[(size_t)q * K_ + k] * 64;
#pragma unroll
    for (int h = 0; h < H_; ++h) {
      float acc = 0.0f;
#pragma unroll
      for (int d = 0; d < 16; ++d) acc += ms[h * 16 + d] * s_src[h * 16 + d];
      float v = acc + sd[h];
      v = (v > 0.0f) ? v : 0.2f * v;                       // leaky_relu(0.2)
      sc[h][k] = v + ebias[(size_t)q * (K_ * H_) + k * H_ + h];
    }
  }

#pragma unroll
  for (int h = 0; h < H_; ++h) {
    float m = -3.4e38f;
#pragma unroll
    for (int k = 0; k < K_; ++k) m = fmaxf(m, sc[h][k]);
    float s = 0.0f;
#pragma unroll
    for (int k = 0; k < K_; ++k) { float e = fast_exp_f(sc[h][k] - m); sc[h][k] = e; s += e; }
    const float inv = 1.0f / s;
#pragma unroll
    for (int k = 0; k < K_; ++k) sc[h][k] *= inv;
  }

  float out[64];
#pragma unroll
  for (int c = 0; c < 64; ++c) out[c] = 0.0f;
#pragma unroll
  for (int k = 0; k < K_; ++k) {
    const float* ms = xp + (size_t)nbr[(size_t)q * K_ + k] * 64;
#pragma unroll
    for (int c = 0; c < 64; ++c) out[c] += sc[c >> 4][k] * ms[c];
  }
  float* nd = nodes + n * 64;
#pragma unroll
  for (int c = 0; c < 64; ++c) nd[c] += out[c];
}

// ---------------------------------------------------------------------------
// Kernel 4: test head: gelu(x@H1+c1) -> gelu(@H2+c2) -> @H3+c3, softplus scale
// ---------------------------------------------------------------------------
constexpr size_t HEAD_H =
    (size_t)WHEAD_H +    // H1t|H2t contiguous panel
    (size_t)64 * 64 +    // A0
    (size_t)64 * 256 +   // act1
    (size_t)64 * 64;     // act2
constexpr size_t HEAD_SMEM = HEAD_H * 2 + (size_t)(256 + 64 + 128 + 2) * 4;

__global__ void __launch_bounds__(128) head_kernel(
    const float* __restrict__ nodes, const _Float16* __restrict__ wbuf,
    const float* __restrict__ c1, const float* __restrict__ c2,
    const float* __restrict__ H3, const float* __restrict__ c3,
    float* __restrict__ out) {
  extern __shared__ char smem[];
  _Float16* sH1  = (_Float16*)smem;            // 256 x 64 (N-major)
  _Float16* sH2  = sH1 + (WB_H2 - WB_H1);      // 64 x 256
  _Float16* sA0  = sH1 + WHEAD_H;              // 64 x 64
  _Float16* sAc1 = sA0 + 64 * 64;              // 64 x 256
  _Float16* sAc2 = sAc1 + 64 * 256;            // 64 x 64
  float* sC1 = (float*)(sAc2 + 64 * 64);
  float* sC2 = sC1 + 256;
  float* sH3 = sC2 + 64;                       // 64 x 2 (f32)
  float* sC3 = sH3 + 128;

  const int t = threadIdx.x;
  {
    const v8h* src = (const v8h*)(wbuf + WB_H1);
    v8h* dst = (v8h*)sH1;
    for (int i = t; i < WHEAD_H / 8; i += 128) dst[i] = src[i];
  }
  for (int i = t; i < 256; i += 128) sC1[i] = c1[i];
  if (t < 64) sC2[t] = c2[t];
  if (t < 128) sH3[t] = H3[t];
  if (t < 2) sC3[t] = c3[t];

  const int base = blockIdx.x * 64;            // test-node tile
  for (int i = t; i < 64 * 64; i += 128) {
    const int r = i >> 6, k = i & 63;
    sA0[i] = (_Float16)nodes[((size_t)NCTX + base + r) * 64 + k];
  }
  __syncthreads();

  const int wave = t >> 5, l = t & 31;
  const int m0 = wave * 16, lane_n = l & 15, mbase = (l >> 4) << 3;

  // layer 1: 16x64 @ 64x256
  {
    v16h a0 = load_A_f16(sA0 + m0 * 64, 64);
    v16h a1 = load_A_f16(sA0 + m0 * 64 + 32, 64);
#pragma unroll
    for (int nt = 0; nt < 16; ++nt) {
      v8f c = {};
      c = WMMA_F16(a0, load_B_f16(sH1 + (nt * 16) * 64, 64), c);
      c = WMMA_F16(a1, load_B_f16(sH1 + (nt * 16) * 64 + 32, 64), c);
      const int n = nt * 16 + lane_n;
      const float bias = sC1[n];
#pragma unroll
      for (int v = 0; v < 8; ++v)
        sAc1[(m0 + mbase + v) * 256 + n] = (_Float16)gelu_f(c[v] + bias);
    }
  }
  __syncthreads();

  // layer 2: 16x256 @ 256x64
  {
    v16h a1[8];
#pragma unroll
    for (int kc = 0; kc < 8; ++kc) a1[kc] = load_A_f16(sAc1 + m0 * 256 + kc * 32, 256);
#pragma unroll
    for (int nt = 0; nt < 4; ++nt) {
      v8f c = {};
#pragma unroll
      for (int kc = 0; kc < 8; ++kc)
        c = WMMA_F16(a1[kc], load_B_f16(sH2 + (nt * 16) * 256 + kc * 32, 256), c);
      const int n = nt * 16 + lane_n;
      const float bias = sC2[n];
#pragma unroll
      for (int v = 0; v < 8; ++v)
        sAc2[(m0 + mbase + v) * 64 + n] = (_Float16)gelu_f(c[v] + bias);
    }
  }
  __syncthreads();

  // layer 3: 64 -> 2, one thread per row; out[0]=mean, out[1]=softplus
  if (t < 64) {
    float o0 = sC3[0], o1 = sC3[1];
#pragma unroll
    for (int k = 0; k < 64; ++k) {
      const float a = (float)sAc2[t * 64 + k];
      o0 += a * sH3[k * 2 + 0];
      o1 += a * sH3[k * 2 + 1];
    }
    const size_t q = (size_t)(base + t);
    out[q * 2 + 0] = o0;
    out[q * 2 + 1] = softplus_f(o1);
  }
}

// ---------------------------------------------------------------------------
// Launcher
// ---------------------------------------------------------------------------
extern "C" void kernel_launch(void* const* d_in, const int* in_sizes, int n_in,
                              void* d_out, int out_size, void* d_ws, size_t ws_size,
                              hipStream_t stream) {
  (void)in_sizes; (void)n_in; (void)out_size; (void)ws_size;
  const float* s_ctx  = (const float*)d_in[0];
  const float* f_ctx  = (const float*)d_in[1];
  const float* s_test = (const float*)d_in[2];
  const float* emb    = (const float*)d_in[3];
  const float* W1 = (const float*)d_in[4];   const float* b1 = (const float*)d_in[5];
  const float* W2 = (const float*)d_in[6];   const float* b2 = (const float*)d_in[7];
  const float* W3 = (const float*)d_in[8];   const float* b3 = (const float*)d_in[9];
  const float* ln_g = (const float*)d_in[10]; const float* ln_b = (const float*)d_in[11];
  const float* Wg = (const float*)d_in[12];
  const float* a_src = (const float*)d_in[13]; const float* a_dst = (const float*)d_in[14];
  const float* We = (const float*)d_in[15];  const float* be = (const float*)d_in[16];
  const float* H1 = (const float*)d_in[17];  const float* c1 = (const float*)d_in[18];
  const float* H2 = (const float*)d_in[19];  const float* c2 = (const float*)d_in[20];
  const float* H3 = (const float*)d_in[21];  const float* c3 = (const float*)d_in[22];

  char* ws = (char*)d_ws;
  float* nodes = (float*)ws;                    ws += (size_t)NTOT * 64 * sizeof(float);
  float* xp    = (float*)ws;                    ws += (size_t)NTOT * 64 * sizeof(float);
  int*   nbr   = (int*)ws;                      ws += (size_t)NTEST * K_ * sizeof(int);
  float* ebias = (float*)ws;                    ws += (size_t)NTEST * K_ * H_ * sizeof(float);
  _Float16* wbuf = (_Float16*)ws;               ws += (size_t)WB_TOT * sizeof(_Float16);

  // >64KB dynamic LDS opt-in (idempotent; deterministic every call)
  (void)hipFuncSetAttribute((const void*)encoder_kernel,
                            hipFuncAttributeMaxDynamicSharedMemorySize, (int)ENC_SMEM);
  (void)hipFuncSetAttribute((const void*)head_kernel,
                            hipFuncAttributeMaxDynamicSharedMemorySize, (int)HEAD_SMEM);

  prep_weights<<<(WB_TOT + 255) / 256, 256, 0, stream>>>(W1, W2, W3, Wg, H1, H2, wbuf);
  knn_kernel<<<NTEST / 256, 256, 0, stream>>>(s_ctx, s_test, We, be, nbr, ebias);
  encoder_kernel<<<NTOT / 64, 128, ENC_SMEM, stream>>>(
      s_ctx, f_ctx, s_test, emb, wbuf, b1, b2, b3, ln_g, ln_b, nodes, xp);
  gat_kernel<<<NTEST / 128, 128, 0, stream>>>(xp, nbr, ebias, a_src, a_dst, nodes);
  head_kernel<<<NTEST / 64, 128, HEAD_SMEM, stream>>>(
      nodes, wbuf, c1, c2, H3, c3, (float*)d_out);
}
